// Net_52587579572323
// MI455X (gfx1250) — compile-verified, matrix operand order checked by reading, not run on hardware
//
#include <hip/hip_runtime.h>
#include <hip/hip_bf16.h>
#include <math.h>

typedef __attribute__((ext_vector_type(16))) _Float16 v16h;
typedef __attribute__((ext_vector_type(8)))  _Float16 half8;
typedef __attribute__((ext_vector_type(8)))  float    v8f;

#define HID 64
#define KNN 8
#define BATCH 8

__device__ __forceinline__ float eluf(float x) { return x > 0.f ? x : (expf(x) - 1.f); }

// Order-preserving float->uint map (for atomicMax-based segment max)
__device__ __forceinline__ unsigned fmap(float f) {
    unsigned b = __float_as_uint(f);
    return (b & 0x80000000u) ? ~b : (b | 0x80000000u);
}
__device__ __forceinline__ float funmap(unsigned u) {
    return __uint_as_float((u & 0x80000000u) ? (u & 0x7fffffffu) : ~u);
}

__device__ __forceinline__ v16h mk16(half8 lo, half8 hi) {
    v16h r;
#pragma unroll
    for (int i = 0; i < 8; ++i) { r[i] = lo[i]; r[8 + i] = hi[i]; }
    return r;
}

__device__ __forceinline__ v8f wmma16(v16h a, v16h b, v8f c) {
    // (neg_a, A, neg_b, B, c_mod, C, reuse_a, reuse_b)
    return __builtin_amdgcn_wmma_f32_16x16x32_f16(false, a, false, b, (short)0, c, false, false);
}

// ---------------------------------------------------------------------------
// Weight prep: transpose conv weights to [out, in] and convert to f16
// ---------------------------------------------------------------------------
__global__ void prep_weights(const float* __restrict__ wc1a, const float* __restrict__ wc1b,
                             const float* __restrict__ wc2a, const float* __restrict__ wc2b,
                             _Float16* __restrict__ w1T1, _Float16* __restrict__ w2T1,
                             _Float16* __restrict__ w1T2, _Float16* __restrict__ w2T2) {
    int idx = blockIdx.x * blockDim.x + threadIdx.x;
    const int n1 = 96 * 128, n2 = 64 * 96;
    if (idx < n1) { int o = idx / 128, i = idx % 128; w1T1[idx] = (_Float16)wc1a[i * 96 + o]; return; }
    idx -= n1;
    if (idx < n2) { int o = idx / 96, i = idx % 96; w2T1[idx] = (_Float16)wc1b[i * 64 + o]; return; }
    idx -= n2;
    if (idx < n1) { int o = idx / 128, i = idx % 128; w1T2[idx] = (_Float16)wc2a[i * 96 + o]; return; }
    idx -= n1;
    if (idx < n2) { int o = idx / 96, i = idx % 96; w2T2[idx] = (_Float16)wc2b[i * 64 + o]; return; }
}

// ---------------------------------------------------------------------------
// Input MLP: x*norm -> elu(10->32) -> elu(32->64); emits f16 features + ||h||^2
// ---------------------------------------------------------------------------
__global__ void input_mlp(const float* __restrict__ x, const float* __restrict__ norm,
                          const float* __restrict__ w1, const float* __restrict__ b1,
                          const float* __restrict__ w2, const float* __restrict__ b2,
                          _Float16* __restrict__ hf, float* __restrict__ x2, int total) {
    __shared__ float sw1[10 * 32], sb1[32], sw2[32 * 64], sb2[64], snorm[10];
    for (int i = threadIdx.x; i < 320; i += blockDim.x) sw1[i] = w1[i];
    for (int i = threadIdx.x; i < 32; i += blockDim.x) sb1[i] = b1[i];
    for (int i = threadIdx.x; i < 2048; i += blockDim.x) sw2[i] = w2[i];
    for (int i = threadIdx.x; i < 64; i += blockDim.x) sb2[i] = b2[i];
    for (int i = threadIdx.x; i < 10; i += blockDim.x) snorm[i] = norm[i];
    __syncthreads();
    int idx = blockIdx.x * blockDim.x + threadIdx.x;
    if (idx >= total) return;
    float xin[10];
#pragma unroll
    for (int i = 0; i < 10; ++i) xin[i] = x[idx * 10 + i] * snorm[i];
    float h1[32];
#pragma unroll
    for (int o = 0; o < 32; ++o) {
        float acc = sb1[o];
#pragma unroll
        for (int i = 0; i < 10; ++i) acc += xin[i] * sw1[i * 32 + o];
        h1[o] = eluf(acc);
    }
    float s2 = 0.f;
    for (int o = 0; o < HID; ++o) {
        float acc = sb2[o];
#pragma unroll
        for (int i = 0; i < 32; ++i) acc += h1[i] * sw2[i * HID + o];
        float v = eluf(acc);
        s2 += v * v;
        hf[(size_t)idx * HID + o] = (_Float16)v;
    }
    x2[idx] = s2;
}

__global__ void fill_kernel(unsigned* __restrict__ p, unsigned v, size_t n) {
    size_t i = (size_t)blockIdx.x * blockDim.x + threadIdx.x;
    if (i < n) p[i] = v;
}

// ---------------------------------------------------------------------------
// kNN: one wave = 16 queries; Gram via v_wmma_f32_16x16x32_f16; top-8 via
// LDS transpose + per-lane branchless worst-slot replacement.
// ---------------------------------------------------------------------------
__global__ __launch_bounds__(256) void knn_kernel(const _Float16* __restrict__ hf,
                                                  const float* __restrict__ x2,
                                                  int* __restrict__ nbr, int N) {
    __shared__ float sdist[8][16][17];
    const int lane = threadIdx.x & 31;
    const int wave = threadIdx.x >> 5;
    const int gw = blockIdx.x * (blockDim.x >> 5) + wave;
    const int tilesPerBatch = N / 16;
    const int b = gw / tilesPerBatch;
    const int q0 = (gw % tilesPerBatch) * 16;
    const int mrow = lane & 15;
    const int rb = (lane < 16) ? 0 : 8;   // K-chunk base for A/B, and C row base

    const _Float16* qrow = hf + (size_t)(b * N + q0 + mrow) * HID;
    const v16h A0 = mk16(*(const half8*)(qrow + rb), *(const half8*)(qrow + rb + 16));
    const v16h A1 = mk16(*(const half8*)(qrow + rb + 32), *(const half8*)(qrow + rb + 48));

    float x2q[8];
#pragma unroll
    for (int r = 0; r < 8; ++r) x2q[r] = x2[b * N + q0 + rb + r];

    float bd[8]; int bi[8];
#pragma unroll
    for (int r = 0; r < 8; ++r) { bd[r] = 3e38f; bi[r] = 0; }

    for (int j0 = 0; j0 < N; j0 += 16) {
        const _Float16* crow = hf + (size_t)(b * N + j0 + mrow) * HID;
        const v16h B0 = mk16(*(const half8*)(crow + rb), *(const half8*)(crow + rb + 16));
        const v16h B1 = mk16(*(const half8*)(crow + rb + 32), *(const half8*)(crow + rb + 48));
        v8f acc = {0.f, 0.f, 0.f, 0.f, 0.f, 0.f, 0.f, 0.f};
        acc = wmma16(A0, B0, acc);
        acc = wmma16(A1, B1, acc);
        const float x2c = x2[b * N + j0 + mrow];
#pragma unroll
        for (int r = 0; r < 8; ++r) {
            const int m = rb + r;
            float d = x2q[r] + x2c - 2.f * acc[r];
            if (q0 + m == j0 + mrow) d = 1e9f;   // exclude self
            sdist[wave][m][mrow] = d;
        }
        if (lane < 16) {
#pragma unroll
            for (int c = 0; c < 16; ++c) {
                const float d = sdist[wave][lane][c];
                const int j = j0 + c;
                int ws = 0; float wd = bd[0];
#pragma unroll
                for (int s = 1; s < 8; ++s) { bool g = bd[s] > wd; wd = g ? bd[s] : wd; ws = g ? s : ws; }
                if (d < wd) { bd[ws] = d; bi[ws] = j; }
            }
        }
    }
    if (lane < 16) {
        const int q = q0 + lane;
#pragma unroll
        for (int s = 0; s < 8; ++s) nbr[(size_t)(b * N + q) * KNN + s] = bi[s];
    }
}

// ---------------------------------------------------------------------------
// Fused EdgeConv: one wave = 16 symmetrized edges.
// GEMM1 (K=128 -> 96) ELU -> LDS transpose -> GEMM2 (K=96 -> 64) ELU ->
// segment max via uint-mapped atomicMax.
// ---------------------------------------------------------------------------
__global__ __launch_bounds__(128) void edgeconv_kernel(
    const _Float16* __restrict__ hf, const int* __restrict__ nbr,
    const _Float16* __restrict__ w1T, const float* __restrict__ b1,
    const _Float16* __restrict__ w2T, const float* __restrict__ b2,
    unsigned* __restrict__ outU, int N) {
    __shared__ __align__(16) _Float16 sw1[96 * 128];
    __shared__ __align__(16) _Float16 sw2[64 * 96];
    __shared__ float sb1[96], sb2[64];
    __shared__ __align__(16) _Float16 tbuf[4][16 * 96];
    __shared__ int sdst[4][16];

    for (int i = threadIdx.x; i < 96 * 128; i += blockDim.x) sw1[i] = w1T[i];
    for (int i = threadIdx.x; i < 64 * 96; i += blockDim.x) sw2[i] = w2T[i];
    for (int i = threadIdx.x; i < 96; i += blockDim.x) sb1[i] = b1[i];
    for (int i = threadIdx.x; i < 64; i += blockDim.x) sb2[i] = b2[i];
    __syncthreads();

    const int lane = threadIdx.x & 31;
    const int wave = threadIdx.x >> 5;
    const int tilesPerBatch = (2 * N * KNN) / 16;   // == N for KNN=8
    const int gw = blockIdx.x * 4 + wave;
    const int b = gw / tilesPerBatch;
    const int e0 = (gw % tilesPerBatch) * 16;
    const int mrow = lane & 15;
    const int c0 = (lane < 16) ? 0 : 8;
    const int rb8 = c0;
    const int NK = N * KNN;

    // Edge for my assigned row (forward then reversed edges)
    const int e = e0 + mrow;
    int dst, src;
    if (e < NK) { dst = e >> 3; src = nbr[(size_t)(b * N + dst) * KNN + (e & 7)]; }
    else { const int e2 = e - NK; const int i = e2 >> 3; src = i; dst = nbr[(size_t)(b * N + i) * KNN + (e2 & 7)]; }
    if (lane < 16) sdst[wave][mrow] = dst;

    const _Float16* xd = hf + (size_t)(b * N + dst) * HID;
    const _Float16* xs = hf + (size_t)(b * N + src) * HID;

    // A tiles: K 0..63 = x_d ; K 64..127 = x_s - x_d
    v16h A[4];
    A[0] = mk16(*(const half8*)(xd + c0), *(const half8*)(xd + c0 + 16));
    A[1] = mk16(*(const half8*)(xd + c0 + 32), *(const half8*)(xd + c0 + 48));
    {
        const half8 d0 = *(const half8*)(xs + c0) - *(const half8*)(xd + c0);
        const half8 d1 = *(const half8*)(xs + c0 + 16) - *(const half8*)(xd + c0 + 16);
        const half8 d2 = *(const half8*)(xs + c0 + 32) - *(const half8*)(xd + c0 + 32);
        const half8 d3 = *(const half8*)(xs + c0 + 48) - *(const half8*)(xd + c0 + 48);
        A[2] = mk16(d0, d1);
        A[3] = mk16(d2, d3);
    }

    // GEMM1: [16,128] x [128,96]
#pragma unroll
    for (int nt = 0; nt < 6; ++nt) {
        const int ncol = nt * 16 + mrow;
        const float bias = sb1[ncol];
        v8f acc = {bias, bias, bias, bias, bias, bias, bias, bias};
        const _Float16* wrow = sw1 + ncol * 128;
#pragma unroll
        for (int ks = 0; ks < 4; ++ks) {
            const int kc = ks * 32 + c0;
            const v16h Bm = mk16(*(const half8*)(wrow + kc), *(const half8*)(wrow + kc + 16));
            acc = wmma16(A[ks], Bm, acc);
        }
#pragma unroll
        for (int r = 0; r < 8; ++r) tbuf[wave][(rb8 + r) * 96 + ncol] = (_Float16)eluf(acc[r]);
    }

    // Transposed activations -> A operand for GEMM2 (K=96)
    const _Float16* arow = tbuf[wave] + mrow * 96;
    v16h A2[3];
#pragma unroll
    for (int ks = 0; ks < 3; ++ks) {
        const int kc = ks * 32 + c0;
        A2[ks] = mk16(*(const half8*)(arow + kc), *(const half8*)(arow + kc + 16));
    }

    // GEMM2: [16,96] x [96,64] -> ELU -> atomic segment max
#pragma unroll
    for (int nt = 0; nt < 4; ++nt) {
        const int ncol = nt * 16 + mrow;
        const float bias = sb2[ncol];
        v8f acc = {bias, bias, bias, bias, bias, bias, bias, bias};
        const _Float16* wrow = sw2 + ncol * 96;
#pragma unroll
        for (int ks = 0; ks < 3; ++ks) {
            const int kc = ks * 32 + c0;
            const v16h Bm = mk16(*(const half8*)(wrow + kc), *(const half8*)(wrow + kc + 16));
            acc = wmma16(A2[ks], Bm, acc);
        }
#pragma unroll
        for (int r = 0; r < 8; ++r) {
            const int dm = sdst[wave][rb8 + r];
            const unsigned u = fmap(eluf(acc[r]));
            atomicMax(&outU[(size_t)(b * N + dm) * HID + ncol], u);
        }
    }
}

// ---------------------------------------------------------------------------
// Pairwise max pool (mapped-uint domain) -> f16 features + ||h||^2
// ---------------------------------------------------------------------------
__global__ void pool_kernel(const unsigned* __restrict__ in, _Float16* __restrict__ outh,
                            float* __restrict__ x2, int halfN, int N) {
    int idx = blockIdx.x * blockDim.x + threadIdx.x;
    if (idx >= BATCH * halfN) return;
    const int b = idx / halfN, i = idx % halfN;
    const unsigned* p = in + (size_t)(b * N + 2 * i) * HID;
    float s2 = 0.f;
    for (int c = 0; c < HID; ++c) {
        const unsigned ua = p[c], ub = p[c + HID];
        const float f = funmap(ua > ub ? ua : ub);
        s2 += f * f;
        outh[(size_t)idx * HID + c] = (_Float16)f;
    }
    x2[idx] = s2;
}

// ---------------------------------------------------------------------------
// Global max pool + output head
// ---------------------------------------------------------------------------
__global__ __launch_bounds__(64) void final_kernel(
    const unsigned* __restrict__ in, const float* __restrict__ w_o1, const float* __restrict__ b_o1,
    const float* __restrict__ w_o2, const float* __restrict__ b_o2,
    const float* __restrict__ w_o3, const float* __restrict__ b_o3,
    float* __restrict__ out, int n) {
    __shared__ float g[64], o1[64], o2[32];
    const int b = blockIdx.x, c = threadIdx.x;
    unsigned u = 0u;
    for (int i = 0; i < n; ++i) {
        const unsigned v = in[(size_t)(b * n + i) * HID + c];
        u = v > u ? v : u;
    }
    g[c] = funmap(u);
    __syncthreads();
    float a1 = b_o1[c];
    for (int i = 0; i < 64; ++i) a1 += g[i] * w_o1[i * 64 + c];
    o1[c] = eluf(a1);
    __syncthreads();
    if (c < 32) {
        float a2 = b_o2[c];
        for (int i = 0; i < 64; ++i) a2 += o1[i] * w_o2[i * 32 + c];
        o2[c] = eluf(a2);
    }
    __syncthreads();
    if (c < 2) {
        float a3 = b_o3[c];
        for (int i = 0; i < 32; ++i) a3 += o2[i] * w_o3[i * 2 + c];
        float r;
        if (c == 0) {
            r = fmaxf(a3, 0.f) + log1pf(expf(-fabsf(a3)));                 // softplus
        } else {
            const float s = 1.f / (1.f + expf(-a3));
            r = 3.14159265358979323846f * (2.f * s - 1.f);                 // pi*(2*sigmoid-1)
        }
        out[b * 2 + c] = r;
    }
}

// ---------------------------------------------------------------------------
extern "C" void kernel_launch(void* const* d_in, const int* in_sizes, int n_in,
                              void* d_out, int out_size, void* d_ws, size_t ws_size,
                              hipStream_t stream) {
    (void)in_sizes; (void)n_in; (void)out_size; (void)ws_size;
    const float* x      = (const float*)d_in[0];
    const float* dnorm  = (const float*)d_in[1];
    const float* w_in1  = (const float*)d_in[2];
    const float* b_in1  = (const float*)d_in[3];
    const float* w_in2  = (const float*)d_in[4];
    const float* b_in2  = (const float*)d_in[5];
    const float* w_c1a  = (const float*)d_in[6];
    const float* b_c1a  = (const float*)d_in[7];
    const float* w_c1b  = (const float*)d_in[8];
    const float* b_c1b  = (const float*)d_in[9];
    const float* w_c2a  = (const float*)d_in[10];
    const float* b_c2a  = (const float*)d_in[11];
    const float* w_c2b  = (const float*)d_in[12];
    const float* b_c2b  = (const float*)d_in[13];
    const float* w_o1   = (const float*)d_in[14];
    const float* b_o1   = (const float*)d_in[15];
    const float* w_o2   = (const float*)d_in[16];
    const float* b_o2   = (const float*)d_in[17];
    const float* w_o3   = (const float*)d_in[18];
    const float* b_o3   = (const float*)d_in[19];
    float* out = (float*)d_out;

    const int N1 = 4096, N2 = 2048;

    char* p = (char*)d_ws;
    auto alloc = [&](size_t bytes) -> void* {
        void* r = (void*)p;
        p += (bytes + 255) & ~(size_t)255;
        return r;
    };
    _Float16* hf    = (_Float16*)alloc((size_t)BATCH * N1 * HID * 2);
    float*    x2a   = (float*)   alloc((size_t)BATCH * N1 * 4);
    int*      nbr1  = (int*)     alloc((size_t)BATCH * N1 * KNN * 4);
    unsigned* out1  = (unsigned*)alloc((size_t)BATCH * N1 * HID * 4);
    _Float16* hp    = (_Float16*)alloc((size_t)BATCH * N2 * HID * 2);
    float*    x2p   = (float*)   alloc((size_t)BATCH * N2 * 4);
    int*      nbr2  = (int*)     alloc((size_t)BATCH * N2 * KNN * 4);
    unsigned* out2  = (unsigned*)alloc((size_t)BATCH * N2 * HID * 4);
    _Float16* w1T1  = (_Float16*)alloc(96 * 128 * 2);
    _Float16* w2T1  = (_Float16*)alloc(64 * 96 * 2);
    _Float16* w1T2  = (_Float16*)alloc(96 * 128 * 2);
    _Float16* w2T2  = (_Float16*)alloc(64 * 96 * 2);

    const unsigned INIT = 0x3FFFFFFFu;  // fmap(-2.0f); all ELU outputs > -1 beat this

    prep_weights<<<(2 * (96 * 128 + 64 * 96) + 255) / 256, 256, 0, stream>>>(
        w_c1a, w_c1b, w_c2a, w_c2b, w1T1, w2T1, w1T2, w2T2);

    input_mlp<<<(BATCH * N1 + 255) / 256, 256, 0, stream>>>(
        x, dnorm, w_in1, b_in1, w_in2, b_in2, hf, x2a, BATCH * N1);

    fill_kernel<<<(BATCH * N1 * HID + 255) / 256, 256, 0, stream>>>(out1, INIT, (size_t)BATCH * N1 * HID);
    knn_kernel<<<(BATCH * (N1 / 16)) / 8, 256, 0, stream>>>(hf, x2a, nbr1, N1);
    edgeconv_kernel<<<(BATCH * N1) / 4, 128, 0, stream>>>(hf, nbr1, w1T1, b_c1a, w2T1, b_c1b, out1, N1);

    pool_kernel<<<(BATCH * N2 + 255) / 256, 256, 0, stream>>>(out1, hp, x2p, N2, N1);

    fill_kernel<<<(BATCH * N2 * HID + 255) / 256, 256, 0, stream>>>(out2, INIT, (size_t)BATCH * N2 * HID);
    knn_kernel<<<(BATCH * (N2 / 16)) / 8, 256, 0, stream>>>(hp, x2p, nbr2, N2);
    edgeconv_kernel<<<(BATCH * N2) / 4, 128, 0, stream>>>(hp, nbr2, w1T2, b_c2a, w2T2, b_c2b, out2, N2);

    final_kernel<<<BATCH, 64, 0, stream>>>(out2, w_o1, b_o1, w_o2, b_o2, w_o3, b_o3, out, N2);
}
